// ProbSparseAttention_74174085202107
// MI455X (gfx1250) — compile-verified, matrix operand order checked by reading, not run on hardware
//
#include <hip/hip_runtime.h>
#include <math.h>

// ProbSparse attention (Informer) for B=16, L=2048, D=64, C=30 on gfx1250.
// U = min(int(L*ln L), L) == L  =>  the key "sample" is a permutation, so
// M = rowmax(QK^T) - rowmean(QK^T) exactly and the seed is irrelevant.
// u = int(30*ln 2048) = 228 active queries per batch.
//
// Pipeline: [swizzle Q/K] [swizzle V] [fill mean] -> [measure via WMMA]
//           -> [rank-based top-u] -> [flash attention on gathered rows].
// Q/K/V are pre-converted to bf16 and stored in the per-lane WMMA operand
// layout so the hot loops are pure b128 loads + v_wmma.

#define BATCH 16
#define SEQ   2048
#define HDIM  64

typedef __attribute__((ext_vector_type(16))) __bf16 v16bf;
typedef __attribute__((ext_vector_type(8)))  float  v8f;

struct BF16Frag {
  union {
    v16bf v;
    unsigned int   u32[8];
    unsigned short u16[16];
  };
};

__device__ __forceinline__ unsigned short f2bf(float f) {
  unsigned int u = __float_as_uint(f);
  u += 0x7FFFu + ((u >> 16) & 1u);  // round-to-nearest-even
  return (unsigned short)(u >> 16);
}

// Pack two f32 -> packed bf16x2 (hardware cvt if available).
__device__ __forceinline__ unsigned int cvt_pk_bf16(float lo, float hi) {
#if __has_builtin(__builtin_amdgcn_cvt_pk_bf16_f32)
  typedef __attribute__((ext_vector_type(2))) __bf16 v2bf;
  union { v2bf v; unsigned int u; } c;
  c.v = __builtin_amdgcn_cvt_pk_bf16_f32(lo, hi);
  return c.u;
#else
  return (unsigned int)f2bf(lo) | ((unsigned int)f2bf(hi) << 16);
#endif
}

// K-dim column base for 16-bit A/B operand VGPR v, lane-half h (ISA 7.12.2):
// lanes 0-15: V0..3 -> K{0,1..6,7}, V4..7 -> K{16..23}; lanes 16-31: +8
__device__ __forceinline__ int kpair_base(int v, int h) {
  return 2 * v + (v >= 4 ? 8 : 0) + 8 * h;
}

// Build a 16x32 bf16 operand fragment from row-major f32 memory.
// rowptr must already point at this lane's row (row index = lane & 15).
__device__ __forceinline__ BF16Frag load_frag_rows(const float* __restrict__ rowptr,
                                                   int koff) {
  BF16Frag f;
  int lane = threadIdx.x & 31;
  int h = lane >> 4;
#pragma unroll
  for (int v = 0; v < 8; ++v) {
    int c = koff + kpair_base(v, h);
    float2 t = *(const float2*)(rowptr + c);
    f.u32[v] = cvt_pk_bf16(t.x, t.y);
  }
  return f;
}

// Load a pre-swizzled fragment (two b128 loads).
__device__ __forceinline__ BF16Frag load_frag_swz(const unsigned int* __restrict__ p) {
  BF16Frag f;
  *(uint4*)&f.u32[0] = ((const uint4*)p)[0];
  *(uint4*)&f.u32[4] = ((const uint4*)p)[1];
  return f;
}

__device__ __forceinline__ v8f wmma_bf16(const BF16Frag& a, const BF16Frag& b, v8f c) {
  return __builtin_amdgcn_wmma_f32_16x16x32_bf16(
      /*neg_a=*/false, a.v, /*neg_b=*/false, b.v,
      /*c_mod=*/(short)0, c, /*reuse_a=*/false, /*reuse_b=*/false);
}

// Fragment-buffer geometry (units: u32 dwords).
#define FRAG_DW       8                       // 8 dwords per lane per fragment
#define FRAGBLK_DW    (32 * FRAG_DW)          // one fragment for a whole wave
#define QK_TILE_DW    (2 * FRAGBLK_DW)        // 2 k-chunks per 16-row tile
#define V_CHUNK_DW    (4 * FRAGBLK_DW)        // 4 d-tiles per 32-key chunk

// ---------------------------------------------------------------------------
// Kernel 0a: convert+swizzle Q and K into WMMA operand fragments.
// One wave per 16-row tile; wid in [0, 2*BATCH*128).
// ---------------------------------------------------------------------------
__global__ void __launch_bounds__(256)
swizzle_qk_kernel(const float* __restrict__ Q, const float* __restrict__ K,
                  unsigned int* __restrict__ Qswz, unsigned int* __restrict__ Kswz) {
  int wid = blockIdx.x * 8 + (threadIdx.x >> 5);
  int lane = threadIdx.x & 31;
  int mat = wid >> 11;              // 0 = Q, 1 = K
  int tidx = wid & 2047;            // b*128 + tile
  const float* src = (mat ? K : Q) + (size_t)tidx * 16 * HDIM;
  unsigned int* dst = (mat ? Kswz : Qswz) + (size_t)tidx * QK_TILE_DW;
  int r16 = lane & 15;
#pragma unroll
  for (int chunk = 0; chunk < 2; ++chunk) {
    BF16Frag f = load_frag_rows(src + (size_t)r16 * HDIM, chunk * 32);
    unsigned int* d = dst + chunk * FRAGBLK_DW + lane * FRAG_DW;
    ((uint4*)d)[0] = *(uint4*)&f.u32[0];
    ((uint4*)d)[1] = *(uint4*)&f.u32[4];
  }
}

// ---------------------------------------------------------------------------
// Kernel 0b: convert+swizzle V into B-operand fragments (32-key x 16-d tiles).
// One wave per 32-key chunk; wid in [0, BATCH*64).
// ---------------------------------------------------------------------------
__global__ void __launch_bounds__(256)
swizzle_v_kernel(const float* __restrict__ V, unsigned int* __restrict__ Vswz) {
  int wid = blockIdx.x * 8 + (threadIdx.x >> 5);
  int lane = threadIdx.x & 31;
  int b = wid >> 6, chunk = wid & 63;
  int kc = chunk * 32;
  int r16 = lane & 15, h = lane >> 4;
  const float* Vb = V + (size_t)b * SEQ * HDIM;
  unsigned int* dst = Vswz + ((size_t)b * 64 + chunk) * V_CHUNK_DW;
#pragma unroll
  for (int t4 = 0; t4 < 4; ++t4) {
    BF16Frag f;
    int col = t4 * 16 + r16;
#pragma unroll
    for (int v = 0; v < 8; ++v) {
      int kb = kpair_base(v, h);
      f.u32[v] = cvt_pk_bf16(Vb[(size_t)(kc + kb) * HDIM + col],
                             Vb[(size_t)(kc + kb + 1) * HDIM + col]);
    }
    unsigned int* d = dst + t4 * FRAGBLK_DW + lane * FRAG_DW;
    ((uint4*)d)[0] = *(uint4*)&f.u32[0];
    ((uint4*)d)[1] = *(uint4*)&f.u32[4];
  }
}

// ---------------------------------------------------------------------------
// Kernel 1: default output = broadcast of mean(V, axis=-1) per row.
// ---------------------------------------------------------------------------
__global__ void __launch_bounds__(256)
fill_mean_kernel(const float* __restrict__ V, float* __restrict__ out) {
  size_t row = (size_t)blockIdx.x * 256 + threadIdx.x;  // < BATCH*SEQ
  const float4* v4 = (const float4*)(V + row * HDIM);
  float s = 0.f;
#pragma unroll
  for (int i = 0; i < 16; ++i) {
    float4 t = v4[i];
    s += t.x + t.y + t.z + t.w;
  }
  float m = s * (1.0f / (float)HDIM);
  float4 mv = make_float4(m, m, m, m);
  float4* o4 = (float4*)(out + row * HDIM);
#pragma unroll
  for (int i = 0; i < 16; ++i) o4[i] = mv;
}

// ---------------------------------------------------------------------------
// Kernel 2: sparsity measure M[b,i] = max_j(q_i.k_j) - mean_j(q_i.k_j).
// Block = (batch, strip of 16 queries), 8 waves split the 2048 keys.
// Scores as D[key, query] so per-query key-reduction stays within a lane.
// ---------------------------------------------------------------------------
__global__ void __launch_bounds__(256)
probsparse_measure_kernel(const unsigned int* __restrict__ Qswz,
                          const unsigned int* __restrict__ Kswz,
                          float* __restrict__ Mout) {
  __shared__ float smax[8][16];
  __shared__ float ssum[8][16];
  int blk = blockIdx.x;
  int b = blk >> 7, strip = blk & 127;
  int tid = threadIdx.x, w = tid >> 5, lane = tid & 31;
  int r16 = lane & 15, h = lane >> 4;

  const unsigned int* qs = Qswz + (size_t)(b * 128 + strip) * QK_TILE_DW;
  BF16Frag bq0 = load_frag_swz(qs + lane * FRAG_DW);
  BF16Frag bq1 = load_frag_swz(qs + FRAGBLK_DW + lane * FRAG_DW);

  float runmax = -3.0e38f, runsum = 0.f;
  for (int i = 0; i < 16; ++i) {
    int tile = w + 8 * i;  // tiles w, w+8, ..., w+120
    const unsigned int* ks = Kswz + (size_t)(b * 128 + tile) * QK_TILE_DW;
    BF16Frag ak0 = load_frag_swz(ks + lane * FRAG_DW);
    BF16Frag ak1 = load_frag_swz(ks + FRAGBLK_DW + lane * FRAG_DW);
    v8f c = {};
    c = wmma_bf16(ak0, bq0, c);
    c = wmma_bf16(ak1, bq1, c);
#pragma unroll
    for (int j = 0; j < 8; ++j) {  // score(key = tile*16 + j + 8h, query = r16)
      float s = c[j];
      runmax = fmaxf(runmax, s);
      runsum += s;
    }
  }
  runmax = fmaxf(runmax, __shfl_xor(runmax, 16, 32));
  runsum += __shfl_xor(runsum, 16, 32);
  if (h == 0) { smax[w][r16] = runmax; ssum[w][r16] = runsum; }
  __syncthreads();
  if (tid < 16) {
    float mx = smax[0][tid], sm = ssum[0][tid];
#pragma unroll
    for (int ww = 1; ww < 8; ++ww) {
      mx = fmaxf(mx, smax[ww][tid]);
      sm += ssum[ww][tid];
    }
    Mout[(size_t)b * SEQ + strip * 16 + tid] = mx - sm * (1.0f / (float)SEQ);
  }
}

// ---------------------------------------------------------------------------
// Kernel 3: deterministic top-u selection by rank counting.
// rank(i) = #{j : M[j] > M[i] or (M[j]==M[i] and j<i)} == stable descending
// sort position (lax.top_k tie-break); rank<u => idx[rank]=i. No atomics.
// ---------------------------------------------------------------------------
__global__ void __launch_bounds__(256)
topk_select_kernel(const float* __restrict__ M, int* __restrict__ idx, int u) {
  __shared__ float Ml[SEQ];
  int b = blockIdx.x, tid = threadIdx.x;
  const float* Mb = M + (size_t)b * SEQ;
  for (int i = tid; i < SEQ; i += 256) Ml[i] = Mb[i];
  __syncthreads();
  for (int i = tid; i < SEQ; i += 256) {
    float mi = Ml[i];
    int rank = 0;
    for (int j = 0; j < SEQ; ++j) {
      float mj = Ml[j];
      rank += (mj > mi) || (mj == mi && j < i);
    }
    if (rank < u) idx[(size_t)b * u + rank] = i;
  }
}

// ---------------------------------------------------------------------------
// Kernel 4: flash-style attention for the gathered active queries.
// One wave per group of 16 gathered queries. Per 32-key chunk:
//   4 WMMA scores (from pre-swizzled K), butterfly softmax over lane-halves,
//   P staged through 1KB LDS (C-layout -> A-layout), 4 WMMA P.V (pre-swizzled V).
// ---------------------------------------------------------------------------
__global__ void __launch_bounds__(32)
sparse_attn_kernel(const float* __restrict__ Q,
                   const unsigned int* __restrict__ Kswz,
                   const unsigned int* __restrict__ Vswz,
                   const int* __restrict__ idx,
                   float* __restrict__ out, int u) {
  __shared__ unsigned short pbuf[16 * 32];  // probabilities, bf16, [q][key]

  int g = blockIdx.x, b = blockIdx.y;
  int lane = threadIdx.x;
  int r16 = lane & 15, h = lane >> 4;

  const float* Qb = Q + (size_t)b * SEQ * HDIM;

  // A operand rows = gathered queries (row index = lane & 15)
  int posA = g * 16 + r16;
  int qrowA = (posA < u) ? idx[(size_t)b * u + posA] : 0;
  BF16Frag aq0 = load_frag_rows(Qb + (size_t)qrowA * HDIM, 0);
  BF16Frag aq1 = load_frag_rows(Qb + (size_t)qrowA * HDIM, 32);

  // per-accumulator-row query ids + validity for the final scatter
  int  qst[8];
  bool qok[8];
#pragma unroll
  for (int j = 0; j < 8; ++j) {
    int pos = g * 16 + j + 8 * h;
    qok[j] = pos < u;
    qst[j] = qok[j] ? idx[(size_t)b * u + pos] : 0;
  }

  float mrow[8], lrow[8];
  v8f oacc[4] = {v8f{}, v8f{}, v8f{}, v8f{}};
#pragma unroll
  for (int j = 0; j < 8; ++j) { mrow[j] = -3.0e38f; lrow[j] = 0.f; }
  const float scale = 0.125f;  // 1/sqrt(64)

  for (int kc = 0; kc < SEQ; kc += 32) {
    // scores D[q, key] for two 16-key tiles, K fragments pre-swizzled
    const unsigned int* ks = Kswz + (size_t)(b * 128 + (kc >> 4)) * QK_TILE_DW;
    v8f c0 = {}, c1 = {};
    {
      BF16Frag bk = load_frag_swz(ks + lane * FRAG_DW);
      c0 = wmma_bf16(aq0, bk, c0);
      bk = load_frag_swz(ks + FRAGBLK_DW + lane * FRAG_DW);
      c0 = wmma_bf16(aq1, bk, c0);
      bk = load_frag_swz(ks + QK_TILE_DW + lane * FRAG_DW);
      c1 = wmma_bf16(aq0, bk, c1);
      bk = load_frag_swz(ks + QK_TILE_DW + FRAGBLK_DW + lane * FRAG_DW);
      c1 = wmma_bf16(aq1, bk, c1);
    }

    // online softmax per query row (row = 16-lane half)
#pragma unroll
    for (int j = 0; j < 8; ++j) {
      float s0 = c0[j] * scale, s1 = c1[j] * scale;
      float t = fmaxf(s0, s1);
      t = fmaxf(t, __shfl_xor(t, 1, 32));
      t = fmaxf(t, __shfl_xor(t, 2, 32));
      t = fmaxf(t, __shfl_xor(t, 4, 32));
      t = fmaxf(t, __shfl_xor(t, 8, 32));
      float mnew = fmaxf(mrow[j], t);
      float p0 = __expf(s0 - mnew), p1 = __expf(s1 - mnew);
      float ps = p0 + p1;
      ps += __shfl_xor(ps, 1, 32);
      ps += __shfl_xor(ps, 2, 32);
      ps += __shfl_xor(ps, 4, 32);
      ps += __shfl_xor(ps, 8, 32);
      float corr = __expf(mrow[j] - mnew);
      lrow[j] = lrow[j] * corr + ps;
      mrow[j] = mnew;
      oacc[0][j] *= corr; oacc[1][j] *= corr;
      oacc[2][j] *= corr; oacc[3][j] *= corr;
      int q = j + 8 * h;  // C/D layout: VGPR j holds row j / j+8 per lane half
      unsigned int pp = cvt_pk_bf16(p0, p1);
      pbuf[q * 32 + r16]      = (unsigned short)pp;
      pbuf[q * 32 + 16 + r16] = (unsigned short)(pp >> 16);
    }
    __syncthreads();

    // P (16x32, A layout from LDS) x V chunk (32x64, pre-swizzled B operand)
    BF16Frag pa;
#pragma unroll
    for (int v = 0; v < 8; ++v) {
      int c = kpair_base(v, h);
      pa.u32[v] = *(const unsigned int*)&pbuf[r16 * 32 + c];
    }
    const unsigned int* vs = Vswz + ((size_t)b * 64 + (kc >> 5)) * V_CHUNK_DW;
#pragma unroll
    for (int t4 = 0; t4 < 4; ++t4) {
      BF16Frag bv = load_frag_swz(vs + t4 * FRAGBLK_DW + lane * FRAG_DW);
      oacc[t4] = wmma_bf16(pa, bv, oacc[t4]);
    }
    __syncthreads();  // protect pbuf WAR for next chunk
  }

  // normalize and scatter to the selected output rows
#pragma unroll
  for (int j = 0; j < 8; ++j) {
    if (!qok[j]) continue;
    float inv = 1.0f / lrow[j];
    size_t base = ((size_t)b * SEQ + qst[j]) * HDIM + r16;
    out[base]      = oacc[0][j] * inv;
    out[base + 16] = oacc[1][j] * inv;
    out[base + 32] = oacc[2][j] * inv;
    out[base + 48] = oacc[3][j] * inv;
  }
}

// ---------------------------------------------------------------------------
extern "C" void kernel_launch(void* const* d_in, const int* in_sizes, int n_in,
                              void* d_out, int out_size, void* d_ws, size_t ws_size,
                              hipStream_t stream) {
  (void)in_sizes; (void)n_in; (void)out_size; (void)ws_size;
  const float* Q = (const float*)d_in[0];
  const float* K = (const float*)d_in[1];
  const float* V = (const float*)d_in[2];
  // d_in[3] (seed) unused: U == n makes the sample a permutation, so the
  // result is RNG-independent.
  float* out = (float*)d_out;

  const int u = (int)(30.0 * log((double)SEQ));  // 228
  const int groups = (u + 15) / 16;              // 15

  // Workspace layout (bytes):
  //   [0,128K)      M measure        (BATCH*SEQ f32)
  //   [128K,144K)   top-u indices    (BATCH*u i32, padded)
  //   [144K, +4MB)  Qswz  [+4MB) Kswz  [+4MB) Vswz
  char* ws = (char*)d_ws;
  float* Mbuf = (float*)ws;
  int*   idxbf = (int*)(ws + (size_t)BATCH * SEQ * sizeof(float));
  unsigned int* Qswz = (unsigned int*)(ws + 147456);
  unsigned int* Kswz = Qswz + (size_t)BATCH * 128 * QK_TILE_DW;
  unsigned int* Vswz = Kswz + (size_t)BATCH * 128 * QK_TILE_DW;

  swizzle_qk_kernel<<<2 * BATCH * 128 / 8, 256, 0, stream>>>(Q, K, Qswz, Kswz);
  swizzle_v_kernel<<<BATCH * 64 / 8, 256, 0, stream>>>(V, Vswz);
  fill_mean_kernel<<<BATCH * SEQ / 256, 256, 0, stream>>>(V, out);
  probsparse_measure_kernel<<<BATCH * 128, 256, 0, stream>>>(Qswz, Kswz, Mbuf);
  topk_select_kernel<<<BATCH, 256, 0, stream>>>(Mbuf, idxbf, u);
  dim3 g3(groups, BATCH);
  sparse_attn_kernel<<<g3, 32, 0, stream>>>(Q, Kswz, Vswz, idxbf, out, u);
}
